// Critic_71811853189359
// MI455X (gfx1250) — matrix operand order repaired and checked by reading, not compiled, not run on hardware
//
#include <hip/hip_runtime.h>
#include <hip/hip_bf16.h>
#include <cstdint>
#include <cstddef>

typedef __attribute__((ext_vector_type(16))) _Float16 v16h;
typedef __attribute__((ext_vector_type(8)))  _Float16 v8h;
typedef __attribute__((ext_vector_type(8)))  float    v8f;

__device__ __forceinline__ float lrelu(float v) { return v > 0.f ? v : 0.01f * v; }

// ---------------------------------------------------------------------------
// Small scalar kernels
// ---------------------------------------------------------------------------

__global__ void k_count(const int* __restrict__ dst, float* __restrict__ cnt, int E) {
  int e = blockIdx.x * blockDim.x + threadIdx.x;
  if (e < E) atomicAdd(&cnt[dst[e]], 1.0f);
}

// g[e,k] = leaky_relu(ea[e,:4] @ Wa[:,k] + ba[k])
__global__ void k_edge_hidden(const float* __restrict__ ea, const float* __restrict__ Wa,
                              const float* __restrict__ ba, float* __restrict__ g,
                              int E, int H) {
  int t = blockIdx.x * blockDim.x + threadIdx.x;
  if (t >= E * H) return;
  int e = t / H, k = t - e * H;
  const float* a = ea + (size_t)e * 4;
  float v = ba[k] + a[0] * Wa[0 * H + k] + a[1] * Wa[1 * H + k]
                  + a[2] * Wa[2 * H + k] + a[3] * Wa[3 * H + k];
  g[t] = lrelu(v);
}

// xs[e,i] = din[src[e], i]
__global__ void k_gather(const float* __restrict__ din, const int* __restrict__ src,
                         float* __restrict__ xs, int E, int I) {
  int t = blockIdx.x * blockDim.x + threadIdx.x;
  if (t >= E * I) return;
  int e = t / I, i = t - e * I;
  xs[t] = din[(size_t)src[e] * I + i];
}

// s[dst[e],o] += sum_i xs[e,i] * bb[i*O+o]   (edge-MLP bias term, scattered)
__global__ void k_bias_msg(const float* __restrict__ xs, const float* __restrict__ bb,
                           const int* __restrict__ dst, float* __restrict__ s,
                           int E, int I, int O) {
  int t = blockIdx.x * blockDim.x + threadIdx.x;
  if (t >= E * O) return;
  int e = t / O, o = t - e * O;
  const float* xr = xs + (size_t)e * I;
  float acc = 0.f;
  for (int i = 0; i < I; ++i) acc += xr[i] * bb[(size_t)i * O + o];
  atomicAdd(&s[(size_t)dst[e] * O + o], acc);
}

// ---------------------------------------------------------------------------
// Fused bilinear WMMA kernel with scattered accumulation:
//   s[dst[e], o0+n] += sum_{i in split} sum_{kh} g[e,kh]*xs[e,i]*Wb[kh, i*O+o0+n]
// K flattened as (i, k_hidden): each 32-wide K chunk has constant i, so the
// A-fragment is xs[e,i] * g[e, kh:kh+32] (built in registers) and the B-tile
// is a plain 2D slice of Wb streamed fp32 -> f16 into double-buffered LDS.
// Block = 256 threads = 8 waves sharing the B tile; wave w owns M-tile
// (blockIdx.z*8+w) x 64 output columns (4 v8f accumulators).
// ---------------------------------------------------------------------------

__global__ __launch_bounds__(256)
void k_bilinear(const float* __restrict__ g,   // [E,H]
                const float* __restrict__ xs,  // [E,I]
                const float* __restrict__ Wb,  // [H, I*O]
                const int*   __restrict__ dst, // [E]
                float* __restrict__ s,         // [N,O] atomically accumulated
                int E, int H, int I, int O, int iPerSplit, int lgKh)
{
  // Double-buffered B tile, transposed: Bt[buf][n][k]; row stride 40 halves
  // (80B) keeps v8h loads 16B-aligned and spreads LDS banks.
  __shared__ __align__(16) _Float16 Bt[2][64][40];

  const int lane = threadIdx.x & 31;
  const int wave = threadIdx.x >> 5;
  const int o0   = blockIdx.x * 64;
  const int i0   = blockIdx.y * iPerSplit;
  const int iend = min(i0 + iPerSplit, I);
  const int mtile = blockIdx.z * 8 + wave;
  const int row  = lane & 15;   // M for A-frag / N for B-frag & D
  const int sel  = lane >> 4;   // K-half selector
  const int eA   = min(mtile * 16 + row, E - 1);
  const size_t ldW = (size_t)I * O;
  const int nKh   = H >> 5;
  const int khMask = nKh - 1;
  const int nChunks = (iend - i0) << lgKh;   // flattened (i, kc) loop

  // cooperative B-load lane mapping
  const int kr = threadIdx.x >> 3;        // 0..31 (K row)
  const int oc = (threadIdx.x & 7) << 3;  // 0..56 (col group)

  const float* gp = g + (size_t)eA * H;

  v8f c0 = {}, c1 = {}, c2 = {}, c3 = {};
  float4 w0, w1;                           // staged global B data

  auto loadB = [&](int c) {
    const int i  = i0 + (c >> lgKh);
    const int kb = (c & khMask) << 5;
    const float* bp = Wb + (size_t)i * O + o0 + (size_t)(kb + kr) * ldW + oc;
    w0 = *(const float4*)(bp);
    w1 = *(const float4*)(bp + 4);
    __builtin_prefetch(bp + 32 * ldW, 0, 1);   // speculative: next K row block
  };
  auto storeB = [&](int buf) {
    Bt[buf][oc + 0][kr] = (_Float16)w0.x;  Bt[buf][oc + 1][kr] = (_Float16)w0.y;
    Bt[buf][oc + 2][kr] = (_Float16)w0.z;  Bt[buf][oc + 3][kr] = (_Float16)w0.w;
    Bt[buf][oc + 4][kr] = (_Float16)w1.x;  Bt[buf][oc + 5][kr] = (_Float16)w1.y;
    Bt[buf][oc + 6][kr] = (_Float16)w1.z;  Bt[buf][oc + 7][kr] = (_Float16)w1.w;
  };

  // prologue: fill buffer 0
  loadB(0);
  storeB(0);
  __syncthreads();

  for (int c = 0; c < nChunks; ++c) {
    const int buf = c & 1;
    // issue next chunk's global loads early (hidden under this chunk's math)
    if (c + 1 < nChunks) loadB(c + 1);

    // ---- A fragment: a[m,k] = xs[e,i] * g[e, kb+k] ----
    // A 16x32, contiguous K order (matches B): elem j -> K=sel*16+j
    const int i  = i0 + (c >> lgKh);
    const int kb = (c & khMask) << 5;
    const float xv = xs[(size_t)eA * I + i];
    v16h a;
    {
      const float* ap = gp + kb + sel * 16;
      float4 a0 = *(const float4*)(ap);
      float4 a1 = *(const float4*)(ap + 4);
      float4 a2 = *(const float4*)(ap + 8);
      float4 a3 = *(const float4*)(ap + 12);
      a[0]  = (_Float16)(xv * a0.x); a[1]  = (_Float16)(xv * a0.y);
      a[2]  = (_Float16)(xv * a0.z); a[3]  = (_Float16)(xv * a0.w);
      a[4]  = (_Float16)(xv * a1.x); a[5]  = (_Float16)(xv * a1.y);
      a[6]  = (_Float16)(xv * a1.z); a[7]  = (_Float16)(xv * a1.w);
      a[8]  = (_Float16)(xv * a2.x); a[9]  = (_Float16)(xv * a2.y);
      a[10] = (_Float16)(xv * a2.z); a[11] = (_Float16)(xv * a2.w);
      a[12] = (_Float16)(xv * a3.x); a[13] = (_Float16)(xv * a3.y);
      a[14] = (_Float16)(xv * a3.z); a[15] = (_Float16)(xv * a3.w);
    }

    // ---- preload ALL four B fragments, then WMMAs back-to-back ----
    v16h b0, b1, b2, b3;
    {
      const int ko = sel * 16;
      const v8h l0 = *(const v8h*)&Bt[buf][ 0 + row][ko];
      const v8h h0 = *(const v8h*)&Bt[buf][ 0 + row][ko + 8];
      const v8h l1 = *(const v8h*)&Bt[buf][16 + row][ko];
      const v8h h1 = *(const v8h*)&Bt[buf][16 + row][ko + 8];
      const v8h l2 = *(const v8h*)&Bt[buf][32 + row][ko];
      const v8h h2 = *(const v8h*)&Bt[buf][32 + row][ko + 8];
      const v8h l3 = *(const v8h*)&Bt[buf][48 + row][ko];
      const v8h h3 = *(const v8h*)&Bt[buf][48 + row][ko + 8];
#pragma unroll
      for (int j = 0; j < 8; ++j) {
        b0[j] = l0[j]; b0[8 + j] = h0[j];
        b1[j] = l1[j]; b1[8 + j] = h1[j];
        b2[j] = l2[j]; b2[8 + j] = h2[j];
        b3[j] = l3[j]; b3[8 + j] = h3[j];
      }
    }

    // stage next chunk into the alternate LDS buffer (no reader conflict:
    // its last readers finished before the barrier that ended chunk c-1)
    if (c + 1 < nChunks) storeB(buf ^ 1);

    c0 = __builtin_amdgcn_wmma_f32_16x16x32_f16(false, a, false, b0, (short)0, c0, false, false);
    c1 = __builtin_amdgcn_wmma_f32_16x16x32_f16(false, a, false, b1, (short)0, c1, false, false);
    c2 = __builtin_amdgcn_wmma_f32_16x16x32_f16(false, a, false, b2, (short)0, c2, false, false);
    c3 = __builtin_amdgcn_wmma_f32_16x16x32_f16(false, a, false, b3, (short)0, c3, false, false);

    __syncthreads();
  }

  // ---- epilogue: D layout m = r + 8*sel, n = lane&15; scatter into s[dst] ----
  v8f accs[4] = {c0, c1, c2, c3};
#pragma unroll
  for (int r = 0; r < 8; ++r) {
    int m = mtile * 16 + sel * 8 + r;
    if (m < E) {
      size_t base = (size_t)dst[m] * O + o0 + row;
      atomicAdd(&s[base +  0], accs[0][r]);
      atomicAdd(&s[base + 16], accs[1][r]);
      atomicAdd(&s[base + 32], accs[2][r]);
      atomicAdd(&s[base + 48], accs[3][r]);
    }
  }
}

// ---------------------------------------------------------------------------
// Combine / pooling / FC head
// ---------------------------------------------------------------------------

// dout[n, :O] = lrelu(s/cnt + din@root + bias); dout[n, O:O+F] = x[n,:F]
__global__ void k_node_combine(const float* __restrict__ s, const float* __restrict__ cnt,
                               const float* __restrict__ din, const float* __restrict__ root,
                               const float* __restrict__ bias, const float* __restrict__ x,
                               float* __restrict__ dout, int N, int I, int O, int F) {
  int W = O + F;
  int t = blockIdx.x * blockDim.x + threadIdx.x;
  if (t >= N * W) return;
  int n = t / W, c = t - n * W;
  if (c < O) {
    float v = s[(size_t)n * O + c] / fmaxf(cnt[n], 1.f);
    const float* dr = din + (size_t)n * I;
    for (int i = 0; i < I; ++i) v += dr[i] * root[(size_t)i * O + c];
    v += bias[c];
    dout[(size_t)n * W + c] = lrelu(v);
  } else {
    dout[(size_t)n * W + c] = x[(size_t)n * F + (c - O)];
  }
}

__global__ void k_pool(const float* __restrict__ d3, const int* __restrict__ batch,
                       float* __restrict__ pool, float* __restrict__ pcnt, int N, int W) {
  int t = blockIdx.x * blockDim.x + threadIdx.x;
  if (t >= N * W) return;
  int n = t / W, c = t - n * W;
  atomicAdd(&pool[(size_t)batch[n] * W + c], d3[t]);
  if (c == 0) atomicAdd(&pcnt[batch[n]], 1.f);
}

__global__ void k_pool_fin(float* __restrict__ pool, const float* __restrict__ pcnt,
                           int G, int W) {
  int t = blockIdx.x * blockDim.x + threadIdx.x;
  if (t >= G * W) return;
  pool[t] /= fmaxf(pcnt[t / W], 1.f);
}

__global__ void k_fc(const float* __restrict__ in, const float* __restrict__ Wm,
                     const float* __restrict__ b, float* __restrict__ out,
                     int R, int K, int M, int act) {
  int t = blockIdx.x * blockDim.x + threadIdx.x;
  if (t >= R * M) return;
  int r = t / M, o = t - r * M;
  float v = b[o];
  const float* ir = in + (size_t)r * K;
  for (int k = 0; k < K; ++k) v += ir[k] * Wm[(size_t)k * M + o];
  out[t] = act ? lrelu(v) : v;
}

// ---------------------------------------------------------------------------

extern "C" void kernel_launch(void* const* d_in, const int* in_sizes, int n_in,
                              void* d_out, int out_size, void* d_ws, size_t ws_size,
                              hipStream_t stream) {
  const float* x     = (const float*)d_in[0];
  const float* ea    = (const float*)d_in[1];
  const int*   eidx  = (const int*)d_in[2];
  const int*   batch = (const int*)d_in[3];
  const float* Wa1 = (const float*)d_in[4],  *ba1 = (const float*)d_in[5];
  const float* Wb1 = (const float*)d_in[6],  *bb1 = (const float*)d_in[7];
  const float* rt1 = (const float*)d_in[8],  *bi1 = (const float*)d_in[9];
  const float* Wa2 = (const float*)d_in[10], *ba2 = (const float*)d_in[11];
  const float* Wb2 = (const float*)d_in[12], *bb2 = (const float*)d_in[13];
  const float* rt2 = (const float*)d_in[14], *bi2 = (const float*)d_in[15];
  const float* Wa3 = (const float*)d_in[16], *ba3 = (const float*)d_in[17];
  const float* Wb3 = (const float*)d_in[18], *bb3 = (const float*)d_in[19];
  const float* rt3 = (const float*)d_in[20], *bi3 = (const float*)d_in[21];
  const float* f1W = (const float*)d_in[22], *f1b = (const float*)d_in[23];
  const float* f2W = (const float*)d_in[24], *f2b = (const float*)d_in[25];
  const float* f3W = (const float*)d_in[26], *f3b = (const float*)d_in[27];

  const int F0 = 12;
  const int N = in_sizes[0] / F0;   // 1536
  const int E = in_sizes[1] / 4;    // 1536
  const int G = out_size;           // 64
  const int* src = eidx;
  const int* dst = eidx + E;

  // workspace carve (fp32)
  float* w = (float*)d_ws;
  float* g    = w; w += (size_t)E * 512;
  float* xs   = w; w += (size_t)E * 268;
  float* sbuf = w; w += (size_t)N * 512;
  float* cnt  = w; w += N;
  float* d1   = w; w += (size_t)N * 268;
  float* d2   = w; w += (size_t)N * 268;
  float* d3   = w; w += (size_t)N * 524;
  float* pool = w; w += (size_t)G * 524;
  float* pcnt = w; w += G;
  float* fh1  = w; w += (size_t)G * 768;
  float* fh2  = w; w += (size_t)G * 1024;
  (void)n_in; (void)ws_size;

  const int TB = 256;
  hipMemsetAsync(cnt, 0, (size_t)N * sizeof(float), stream);
  k_count<<<(E + TB - 1) / TB, TB, 0, stream>>>(dst, cnt, E);

  auto conv = [&](const float* din, int I, int H, int O,
                  const float* Wa, const float* ba, const float* Wb, const float* bb,
                  const float* root, const float* bias, float* dout, int ips) {
    int T;
    T = E * H;  k_edge_hidden<<<(T + TB - 1) / TB, TB, 0, stream>>>(ea, Wa, ba, g, E, H);
    T = E * I;  k_gather<<<(T + TB - 1) / TB, TB, 0, stream>>>(din, src, xs, E, I);
    hipMemsetAsync(sbuf, 0, (size_t)N * O * sizeof(float), stream);
    T = E * O;  k_bias_msg<<<(T + TB - 1) / TB, TB, 0, stream>>>(xs, bb, dst, sbuf, E, I, O);
    int nKh = H >> 5;
    int lgKh = (nKh == 8) ? 3 : 4;   // H = 256 or 512
    dim3 bg(O / 64, (I + ips - 1) / ips, (E + 127) / 128);
    k_bilinear<<<bg, TB, 0, stream>>>(g, xs, Wb, dst, sbuf, E, H, I, O, ips, lgKh);
    T = N * (O + F0);
    k_node_combine<<<(T + TB - 1) / TB, TB, 0, stream>>>(
        sbuf, cnt, din, root, bias, x, dout, N, I, O, F0);
  };

  conv(x,  12,  256, 256, Wa1, ba1, Wb1, bb1, rt1, bi1, d1, 12);
  conv(d1, 268, 512, 256, Wa2, ba2, Wb2, bb2, rt2, bi2, d2, 67);
  conv(d2, 268, 512, 512, Wa3, ba3, Wb3, bb3, rt3, bi3, d3, 67);

  const int WP = 524;
  hipMemsetAsync(pool, 0, (size_t)G * WP * sizeof(float), stream);
  hipMemsetAsync(pcnt, 0, (size_t)G * sizeof(float), stream);
  k_pool<<<(N * WP + TB - 1) / TB, TB, 0, stream>>>(d3, batch, pool, pcnt, N, WP);
  k_pool_fin<<<(G * WP + TB - 1) / TB, TB, 0, stream>>>(pool, pcnt, G, WP);
  k_fc<<<(G * 768 + TB - 1) / TB, TB, 0, stream>>>(pool, f1W, f1b, fh1, G, WP, 768, 1);
  k_fc<<<(G * 1024 + TB - 1) / TB, TB, 0, stream>>>(fh1, f2W, f2b, fh2, G, 768, 1024, 1);
  k_fc<<<(G + TB - 1) / TB, TB, 0, stream>>>(fh2, f3W, f3b, (float*)d_out, G, 1024, 1, 0);
}
